// KPRNN_41549513622099
// MI455X (gfx1250) — matrix operand order repaired and checked by reading, not compiled, not run on hardware
//
#include <hip/hip_runtime.h>

// ---------------- problem constants (from reference) ----------------
#define BSZ   2
#define SPP   8
#define NIN   16
#define KS    21
#define KK2   441          // KSIZE^2
#define HH    192
#define WW    192
#define HP    194          // padded (SAME halo)
#define H1    190
#define H2    188
#define H3    186
#define HO    172          // cropped output
#define CROP  7
#define EPSV  1e-8f

#define CIP_RNN 160        // 144 padded to x32
#define KP_RNN  (9 * CIP_RNN)   // 1440
#define CIP_128 128
#define KP_128  (9 * CIP_128)   // 1152

// WMMA / descriptor types (CDNA5 / gfx1250)
typedef __attribute__((ext_vector_type(16))) __bf16       v16bf;
typedef __attribute__((ext_vector_type(8)))  float        v8f;
typedef __attribute__((ext_vector_type(4)))  unsigned int v4u;
typedef __attribute__((ext_vector_type(4)))  unsigned int u32x4;
typedef __attribute__((ext_vector_type(8)))  int          i32x8;
typedef __attribute__((ext_vector_type(4)))  int          i32x4;

__device__ __forceinline__ unsigned short f2bf(float f) {
    unsigned int u = __builtin_bit_cast(unsigned int, f);
    unsigned int lsb = (u >> 16) & 1u;
    u += 0x7fffu + lsb;                       // round-to-nearest-even
    return (unsigned short)(u >> 16);
}

// =====================================================================
// Implicit-GEMM 3x3 conv, tap-major reduction, channel-pair-interleaved
// activations ([c/2][y][x][2] bf16 == u32 per pair).  Geometry is a
// template so every stride folds to an immediate:
//   B fragment = 8 global_load_b32 with constant offsets off one base;
//   A fragment = two ds_load_b128 from the TDM-staged LDS weight tile.
// Two v_wmma_f32_16x16x32_bf16 per 32-chunk, chunk loop unrolled x2.
// =====================================================================
template <int HIN, int WIN, int CINP, int HOUT, int WOUT, int LEAKY>
__global__ __launch_bounds__(256) void conv3x3_wmma(
    const unsigned int* __restrict__ in,
    const unsigned short* __restrict__ wpk, const float* __restrict__ bias,
    int CoReal,
    unsigned int* out1, long ob1, long oc1, long or1,   // pair-interleaved
    unsigned int* out2, long ob2, long oc2, long or2,   // pair-interleaved
    float*        out3, long ob3, long oc3, long or3)   // planar f32
{
    constexpr int  KPAD   = 9 * CINP;
    constexpr long HINWIN = (long)HIN * WIN;            // u32 per pair-plane
    constexpr int  HWOUT  = HOUT * WOUT;
    constexpr int  NCI    = CINP >> 5;

    __shared__ unsigned short ldsA[16 * KPAD];          // [co][k]

    const int tid    = threadIdx.x;
    const int wave   = tid >> 5;
    const int lane   = tid & 31;
    const int b      = blockIdx.z;
    const int coTile = blockIdx.y << 4;
    const int p0     = (blockIdx.x * 8 + wave) << 5;    // 32 pixels / wave

    // ---- stage the whole 16 x KPAD weight tile into LDS (once) ----
    constexpr int tileElems = KPAD << 4;                // 16 rows, contiguous
    const unsigned short* wTile = wpk + (long)coTile * KPAD;
#if __has_builtin(__builtin_amdgcn_tensor_load_to_lds) && __has_builtin(__builtin_amdgcn_s_wait_tensorcnt)
    if (wave == 0) {
        const unsigned long long ga = (unsigned long long)wTile;  // byte addr
        const unsigned int ldsByte  = (unsigned int)(unsigned long long)&ldsA[0];
        u32x4 g0;
        g0[0] = 1u;                                   // count=1, user D#
        g0[1] = ldsByte;                              // lds_addr
        g0[2] = (unsigned int)ga;                     // global_addr[31:0]
        g0[3] = (unsigned int)((ga >> 32) & 0x01FFFFFFu) | (2u << 30); // type=2
        i32x8 g1;
        g1[0] = (int)(1u << 16);                      // data_size=1 (2 bytes)
        g1[1] = (int)((unsigned)(tileElems & 0xFFFF) << 16);     // tensor_dim0 lo
        g1[2] = (int)(((unsigned)tileElems >> 16) | (1u << 16)); // td0 hi | td1=1
        g1[3] = (int)((unsigned)(tileElems & 0xFFFF) << 16);     // tile_dim0
        g1[4] = 0;                                    // tile_dim1/2 unused
        g1[5] = (int)tileElems;                       // tensor_dim0_stride lo
        g1[6] = 0;
        g1[7] = 0;
        i32x4 z4 = {0, 0, 0, 0};
        i32x8 z8 = {0, 0, 0, 0, 0, 0, 0, 0};
        __builtin_amdgcn_tensor_load_to_lds(g0, g1, z4, z4, z8, 0);
        __builtin_amdgcn_s_wait_tensorcnt((short)0);
    }
#else
    for (int i = tid * 8; i < tileElems; i += 256 * 8)
        *(v4u*)&ldsA[i] = *(const v4u*)&wTile[i];
#endif
    __syncthreads();

    // ---- per-lane pixel offsets (column = lane&15 of each 16-px tile) ----
    const int colB = lane & 15;
    const int pA  = p0 + colB;          // fragment-0 pixel
    const int pBp = p0 + 16 + colB;     // fragment-1 pixel
    const int yA = pA / WOUT,  xA = pA - yA * WOUT;     // const-divisor magic
    const int yB = pBp / WOUT, xB = pBp - yB * WOUT;
    const long pix0 = (long)yA * WIN + xA;   // u32 (pair) units; invalid
    const long pix1 = (long)yB * WIN + xB;   // pixels covered by tail pad

    const int hiA  = (lane & 16) ? 16 : 0;
    const int hiBp = (lane & 16) ? 8 : 0;   // hi half in pair units
    const int rowA = lane & 15;

    // per-lane B base pointers: everything else is a compile-time offset
    const unsigned int* const bbase =
        in + (long)b * (CINP >> 1) * HINWIN + (long)hiBp * HINWIN;
    const unsigned int* const base0 = bbase + pix0;
    const unsigned int* const base1 = bbase + pix1;

    const unsigned short* ldsRow = &ldsA[rowA * KPAD];

    v8f acc0 = {0.f, 0.f, 0.f, 0.f, 0.f, 0.f, 0.f, 0.f};
    v8f acc1 = {0.f, 0.f, 0.f, 0.f, 0.f, 0.f, 0.f, 0.f};

    for (int t = 0; t < 9; ++t) {
        const int ty = t / 3;
        const int tx = t - ty * 3;
        const long tapOff = (long)ty * WIN + tx;    // pair units
        const unsigned int* q0 = base0 + tapOff;
        const unsigned int* q1 = base1 + tapOff;

        #pragma unroll 2
        for (int cc = 0; cc < NCI; ++cc) {
            const int kk = t * CINP + (cc << 5);

            // ---- A fragment: two contiguous ds_load_b128 ----
            v16bf fa;
            {
                union { v4u q[2]; v16bf v; } u;
                u.q[0] = *(const v4u*)&ldsRow[kk + hiA];
                u.q[1] = *(const v4u*)&ldsRow[kk + hiA + 8];
                fa = u.v;
            }

            // ---- B fragments: 8 b32 loads each, constant plane offsets ----
            union { unsigned int d[8]; v16bf v; } ub0, ub1;
            #pragma unroll
            for (int r = 0; r < 8; ++r)
                ub0.d[r] = q0[(long)r * HINWIN];
            #pragma unroll
            for (int r = 0; r < 8; ++r)
                ub1.d[r] = q1[(long)r * HINWIN];
            q0 += 16 * HINWIN;
            q1 += 16 * HINWIN;

            acc0 = __builtin_amdgcn_wmma_f32_16x16x32_bf16(
                       false, fa, false, ub0.v, (short)0, acc0, false, false);
            acc1 = __builtin_amdgcn_wmma_f32_16x16x32_bf16(
                       false, fa, false, ub1.v, (short)0, acc1, false, false);
        }
    }

    // ---- epilogue: bias + leaky; D layout: VGPR v -> M = v + 8*(lane>=16) ----
    const int mh = (lane & 16) ? 8 : 0;
    #pragma unroll
    for (int g = 0; g < 2; ++g) {
        const v8f accg = g ? acc1 : acc0;
        const int p = g ? pBp : pA;
        if (p < HWOUT) {
            const int y = g ? yB : yA;
            const int x = g ? xB : xA;
            #pragma unroll
            for (int v = 0; v < 8; v += 2) {
                const int co = coTile + mh + v;     // even
                if (co < CoReal) {
                    float r0 = accg[v]     + bias[co];
                    float r1 = accg[v + 1] + bias[co + 1];
                    if (LEAKY) {
                        r0 = (r0 < 0.f) ? 0.01f * r0 : r0;
                        r1 = (r1 < 0.f) ? 0.01f * r1 : r1;
                    }
                    const unsigned int pk =
                        (unsigned int)f2bf(r0) | ((unsigned int)f2bf(r1) << 16);
                    const long cp = co >> 1;
                    if (out1) out1[(long)b * ob1 + cp * oc1 + (long)y * or1 + x] = pk;
                    if (out2) out2[(long)b * ob2 + cp * oc2 + (long)y * or2 + x] = pk;
                    if (out3) {
                        out3[(long)b * ob3 + (long)co * oc3 + (long)y * or3 + x] = r0;
                        if (co + 1 < CoReal)
                            out3[(long)b * ob3 + (long)(co + 1) * oc3 + (long)y * or3 + x] = r1;
                    }
                }
            }
        }
    }
}

// ------- weight repack: (Co,Cin,3,3) f32 -> bf16 [CoPad][t*CinPad+ci] -------
__global__ void pack_w(const float* __restrict__ w, unsigned short* __restrict__ wpk,
                       int Co, int CoPad, int Cin, int CinPad)
{
    const int  Kpad = 9 * CinPad;
    const long tot  = (long)CoPad * Kpad;
    for (long idx = (long)blockIdx.x * blockDim.x + threadIdx.x; idx < tot;
         idx += (long)gridDim.x * blockDim.x) {
        const long co = idx / Kpad;
        const long k  = idx - co * Kpad;
        const long t  = k / CinPad;
        const long ci = k - t * CinPad;
        float v = (co < Co && ci < Cin) ? w[(co * Cin + ci) * 9 + t] : 0.f;
        wpk[idx] = f2bf(v);
    }
}

// ------ per-step feature pack into both conv inputs (pair-interleaved) ------
__global__ void pack_f(const float* __restrict__ feat,
                       unsigned short* __restrict__ rnn_in,   // 80 pair-planes 194^2
                       unsigned short* __restrict__ k1_in,    // 80 pair-planes 192^2
                       int step)
{
    const long tot = (long)BSZ * NIN * HH * WW;
    for (long idx = (long)blockIdx.x * blockDim.x + threadIdx.x; idx < tot;
         idx += (long)gridDim.x * blockDim.x) {
        const int b = (int)(idx / ((long)NIN * HH * WW));
        long rem = idx - (long)b * NIN * HH * WW;
        const int c = (int)(rem / (HH * WW));
        rem -= (long)c * HH * WW;
        const int y = (int)(rem / WW);
        const int x = (int)(rem - (long)y * WW);
        const float v = feat[((((long)b * SPP + step) * NIN + c) * HH + y) * WW + x];
        const unsigned short bv = f2bf(v);
        // rnn input: channels 0..15 -> pair-planes 0..7
        rnn_in[(((long)b * (CIP_RNN / 2) + (c >> 1)) * HP * HP
                + (long)(y + 1) * HP + (x + 1)) * 2 + (c & 1)] = bv;
        // k1 input: channels 128..143 -> pair-planes 64..71
        k1_in[(((long)b * (CIP_RNN / 2) + 64 + (c >> 1)) * HH * WW
                + (long)y * WW + x) * 2 + (c & 1)] = bv;
    }
}

// ------------- per-batch max of k3 logits: two stages -------------
__global__ void max_stage1(const float* __restrict__ k3, int n, float* __restrict__ partial)
{
    const int b = blockIdx.y;
    const float* p = k3 + (long)b * n;
    float m = -3.4e38f;
    for (long i = (long)blockIdx.x * blockDim.x + threadIdx.x; i < n;
         i += (long)gridDim.x * blockDim.x)
        m = fmaxf(m, p[i]);
    __shared__ float sm[256];
    sm[threadIdx.x] = m;
    __syncthreads();
    for (int s = 128; s > 0; s >>= 1) {
        if (threadIdx.x < s) sm[threadIdx.x] = fmaxf(sm[threadIdx.x], sm[threadIdx.x + s]);
        __syncthreads();
    }
    if (threadIdx.x == 0) partial[(long)b * gridDim.x + blockIdx.x] = sm[0];
}

__global__ void max_stage2(const float* __restrict__ partial, int nparts,
                           float* __restrict__ maxk, float* __restrict__ scale, int step)
{
    const int b = blockIdx.x;
    __shared__ float sm[256];
    float m = -3.4e38f;
    for (int i = threadIdx.x; i < nparts; i += 256)
        m = fmaxf(m, partial[(long)b * nparts + i]);
    sm[threadIdx.x] = m;
    __syncthreads();
    for (int s = 128; s > 0; s >>= 1) {
        if (threadIdx.x < s) sm[threadIdx.x] = fmaxf(sm[threadIdx.x], sm[threadIdx.x + s]);
        __syncthreads();
    }
    if (threadIdx.x == 0) {
        const float nm = sm[0];
        if (step == 0) { maxk[b] = nm; scale[b] = 1.f; }
        else {
            const float om = maxk[b];
            const float mx = fmaxf(om, nm);
            maxk[b] = mx;
            scale[b] = __expf(om - mx);
        }
    }
}

// ------------- online-softmax kernel application (441 taps) -------------
__global__ void apply_acc(const float* __restrict__ k3, const float* __restrict__ rad_all,
                          const float* __restrict__ maxk, const float* __restrict__ scale,
                          float* __restrict__ unnorm, float* __restrict__ sumw, int step)
{
    const int tot = BSZ * HO * HO;
    const int idx = blockIdx.x * blockDim.x + threadIdx.x;
    if (idx >= tot) return;
    const int b   = idx / (HO * HO);
    const int rem = idx - b * HO * HO;
    const int y   = rem / HO;
    const int x   = rem - y * HO;

    const float mk = maxk[b];
    const float sc = scale[b];
    const float* kp = k3 + (long)b * KK2 * H3 * H3 + (long)(CROP + y) * H3 + (CROP + x);
    const float* rp = rad_all + (((long)b * SPP + step) * 3) * (HH * WW);

    float ws0 = 0.f, ws1 = 0.f, ws2 = 0.f, ks = 0.f;
    int t = 0;
    for (int dy = 0; dy < KS; ++dy) {
        const float* r0 = rp + (long)(y + dy) * WW + x;
        for (int dx = 0; dx < KS; ++dx, ++t) {
            const float kv = __expf(kp[(long)t * H3 * H3] - mk);
            ks  += kv;
            ws0 += kv * r0[dx];
            ws1 += kv * r0[dx + (long)HH * WW];
            ws2 += kv * r0[dx + 2L * HH * WW];
        }
    }
    const long u0 = (long)b * 3 * HO * HO + rem;
    const long s0 = (long)b * HO * HO + rem;
    if (step == 0) {
        unnorm[u0]                 = ws0;
        unnorm[u0 + (long)HO * HO] = ws1;
        unnorm[u0 + 2L * HO * HO]  = ws2;
        sumw[s0] = ks;
    } else {
        unnorm[u0]                 = unnorm[u0] * sc + ws0;
        unnorm[u0 + (long)HO * HO] = unnorm[u0 + (long)HO * HO] * sc + ws1;
        unnorm[u0 + 2L * HO * HO]  = unnorm[u0 + 2L * HO * HO] * sc + ws2;
        sumw[s0] = sumw[s0] * sc + ks;
    }
}

__global__ void finalize_k(const float* __restrict__ unnorm, const float* __restrict__ sumw,
                           float* __restrict__ out)
{
    const int tot = BSZ * 3 * HO * HO;
    const int idx = blockIdx.x * blockDim.x + threadIdx.x;
    if (idx >= tot) return;
    const int b   = idx / (3 * HO * HO);
    const int rem = idx - b * 3 * HO * HO;
    const int pix = rem % (HO * HO);
    out[idx] = unnorm[idx] / (sumw[(long)b * HO * HO + pix] + EPSV);
}

// =====================================================================
extern "C" void kernel_launch(void* const* d_in, const int* in_sizes, int n_in,
                              void* d_out, int out_size, void* d_ws, size_t ws_size,
                              hipStream_t stream)
{
    (void)in_sizes; (void)n_in; (void)out_size; (void)ws_size;

    const float* feat  = (const float*)d_in[0];
    const float* rad   = (const float*)d_in[1];
    const float* w_rnn = (const float*)d_in[2];
    const float* b_rnn = (const float*)d_in[3];
    const float* w_k1  = (const float*)d_in[4];
    const float* b_k1  = (const float*)d_in[5];
    const float* w_k2  = (const float*)d_in[6];
    const float* b_k2  = (const float*)d_in[7];
    const float* w_k3  = (const float*)d_in[8];
    const float* b_k3  = (const float*)d_in[9];
    float* out = (float*)d_out;

    // ---------------- workspace layout ----------------
    char* base = (char*)d_ws;
    size_t off = 0;
    auto alloc = [&](size_t bytes) -> char* {
        char* p = base + off;
        off = (off + bytes + 255) & ~(size_t)255;
        return p;
    };
    const size_t TAIL = 8192;   // overread pad for unguarded B loads

    unsigned short* wpk_rnn = (unsigned short*)alloc((size_t)128 * KP_RNN * 2);
    unsigned short* wpk_k1  = (unsigned short*)alloc((size_t)128 * KP_RNN * 2);
    unsigned short* wpk_k2  = (unsigned short*)alloc((size_t)128 * KP_128 * 2);
    unsigned short* wpk_k3  = (unsigned short*)alloc((size_t)448 * KP_128 * 2);

    const size_t rnnInElems = (size_t)BSZ * CIP_RNN * HP * HP;
    const size_t k1InElems  = (size_t)BSZ * CIP_RNN * HH * WW;
    unsigned short* in_rnn[2];
    in_rnn[0] = (unsigned short*)alloc(rnnInElems * 2 + TAIL);
    in_rnn[1] = (unsigned short*)alloc(rnnInElems * 2 + TAIL);
    unsigned short* in_k1  = (unsigned short*)alloc(k1InElems * 2 + TAIL);
    unsigned int*   buf_k1 = (unsigned int*)alloc((size_t)BSZ * 128 * H1 * H1 * 2 + TAIL);
    unsigned int*   buf_k2 = (unsigned int*)alloc((size_t)BSZ * 128 * H2 * H2 * 2 + TAIL);
    float*          buf_k3 = (float*)alloc((size_t)BSZ * KK2 * H3 * H3 * 4);
    float* partial = (float*)alloc((size_t)BSZ * 256 * 4);
    float* maxk    = (float*)alloc(BSZ * 4);
    float* scal    = (float*)alloc(BSZ * 4);
    float* unnorm  = (float*)alloc((size_t)BSZ * 3 * HO * HO * 4);
    float* sumw    = (float*)alloc((size_t)BSZ * HO * HO * 4);

    auto cdiv = [](long a, long b) { return (int)((a + b - 1) / b); };
    const dim3 blk(256);

    // zero padded-channel input buffers (halo + initial state + pad ch = 0)
    (void)hipMemsetAsync(in_rnn[0], 0, rnnInElems * 2 + TAIL, stream);
    (void)hipMemsetAsync(in_rnn[1], 0, rnnInElems * 2 + TAIL, stream);
    (void)hipMemsetAsync(in_k1,     0, k1InElems  * 2 + TAIL, stream);

    // pack weights -> bf16 [CoPad][t*CinPad+ci]
    pack_w<<<512, blk, 0, stream>>>(w_rnn, wpk_rnn, 128, 128, 144, CIP_RNN);
    pack_w<<<512, blk, 0, stream>>>(w_k1,  wpk_k1,  128, 128, 144, CIP_RNN);
    pack_w<<<512, blk, 0, stream>>>(w_k2,  wpk_k2,  128, 128, 128, CIP_128);
    pack_w<<<512, blk, 0, stream>>>(w_k3,  wpk_k3,  441, 448, 128, CIP_128);

    for (int step = 0; step < SPP; ++step) {
        const int cur = step & 1;
        const int nxt = cur ^ 1;

        // features -> rnn input (pairs 0..7, haloed) and k1 input (pairs 64..71)
        pack_f<<<512, blk, 0, stream>>>(feat, in_rnn[cur], in_k1, step);

        // RNN conv (SAME, 144->128, leaky): out1 = k1 input pairs 0..63,
        // out2 = next-step state (pairs 8..71 of the *other* haloed buffer)
        conv3x3_wmma<HP, HP, CIP_RNN, HH, WW, 1>
            <<<dim3(cdiv((long)HH * WW, 256), 8, BSZ), blk, 0, stream>>>(
            (const unsigned int*)in_rnn[cur], wpk_rnn, b_rnn, 128,
            (unsigned int*)in_k1, (long)(CIP_RNN / 2) * HH * WW, (long)HH * WW, WW,
            (unsigned int*)in_rnn[nxt] + (long)8 * HP * HP + HP + 1,
            (long)(CIP_RNN / 2) * HP * HP, (long)HP * HP, HP,
            nullptr, 0, 0, 0);

        // k1 conv (VALID, 144->128, leaky)
        conv3x3_wmma<HH, WW, CIP_RNN, H1, H1, 1>
            <<<dim3(cdiv((long)H1 * H1, 256), 8, BSZ), blk, 0, stream>>>(
            (const unsigned int*)in_k1, wpk_k1, b_k1, 128,
            buf_k1, (long)64 * H1 * H1, (long)H1 * H1, H1,
            nullptr, 0, 0, 0, nullptr, 0, 0, 0);

        // k2 conv (VALID, 128->128, leaky)
        conv3x3_wmma<H1, H1, CIP_128, H2, H2, 1>
            <<<dim3(cdiv((long)H2 * H2, 256), 8, BSZ), blk, 0, stream>>>(
            buf_k1, wpk_k2, b_k2, 128,
            buf_k2, (long)64 * H2 * H2, (long)H2 * H2, H2,
            nullptr, 0, 0, 0, nullptr, 0, 0, 0);

        // k3 conv (VALID, 128->441, linear, f32 logits)
        conv3x3_wmma<H2, H2, CIP_128, H3, H3, 0>
            <<<dim3(cdiv((long)H3 * H3, 256), 28, BSZ), blk, 0, stream>>>(
            buf_k2, wpk_k3, b_k3, 441,
            nullptr, 0, 0, 0, nullptr, 0, 0, 0,
            buf_k3, (long)KK2 * H3 * H3, (long)H3 * H3, H3);

        // per-batch running max + rescale factor
        max_stage1<<<dim3(256, BSZ), blk, 0, stream>>>(buf_k3, KK2 * H3 * H3, partial);
        max_stage2<<<BSZ, blk, 0, stream>>>(partial, 256, maxk, scal, step);

        // exp + 441-tap filter + online accumulation
        apply_acc<<<cdiv((long)BSZ * HO * HO, 256), blk, 0, stream>>>(
            buf_k3, rad, maxk, scal, unnorm, sumw, step);
    }

    finalize_k<<<cdiv((long)BSZ * 3 * HO * HO, 256), blk, 0, stream>>>(unnorm, sumw, out);
}
